// LocalTransformerBlock_33028298506630
// MI455X (gfx1250) — matrix operand order repaired and checked by reading, compiled, not run on hardware
//
#include <hip/hip_runtime.h>
#include <hip/hip_bf16.h>

// ---------------- problem constants (from reference) ----------------
#define B_    4
#define L_    2048
#define D_    256
#define H_    8
#define DH_   32
#define MLP_  1024
#define MAXREL_ 16
#define ROWS_ (B_ * L_)        // 8192

typedef __attribute__((ext_vector_type(16))) __bf16 v16bf;
typedef __attribute__((ext_vector_type(8)))  float  v8f;

// pack two floats into two bf16 in one dword: single v_perm_b32
// bytes {a[3:2]} -> out[1:0], {b[3:2]} -> out[3:2]
__device__ __forceinline__ unsigned pk2(float a, float b) {
    return __builtin_amdgcn_perm(__builtin_bit_cast(unsigned, b),
                                 __builtin_bit_cast(unsigned, a),
                                 0x07060302u);
}
__device__ __forceinline__ uint4 pk8(const float* s) {
    return uint4{pk2(s[0], s[1]), pk2(s[2], s[3]), pk2(s[4], s[5]), pk2(s[6], s[7])};
}
__device__ __forceinline__ __bf16 f2bf(float f) {
    unsigned short h = (unsigned short)(__builtin_bit_cast(unsigned, f) >> 16);
    return __builtin_bit_cast(__bf16, h);
}
// fragment load: two 16-byte contiguous runs from LDS -> v16bf (2x ds_load_b128)
__device__ __forceinline__ v16bf ld_frag(const __bf16* p0, const __bf16* p1) {
    union { v16bf v; uint4 q[2]; } u;
    u.q[0] = *(const uint4*)p0;
    u.q[1] = *(const uint4*)p1;
    return u.v;
}

// =====================================================================
// GEMM: C[M,N] = A[M,K] @ B[K,N] + bias, optional ReLU.
// fp32 in/out, bf16 WMMA. Block = 256 thr (8 waves), tile 128x128, K=32.
// Wave (wr,wc) owns a 32x64 sub-tile: 2 A-frags x 4 B-frags = 8 WMMA/iter.
// LDS layouts are fragment-major: As[m][k], BsT[n][k] (both k-contiguous).
// =====================================================================
#define BM 128
#define BN 128
#define BK 32
#define LDK 40   // padded row length (80B, 16B-aligned, bank-spread)

__global__ __launch_bounds__(256) void gemm_bf16(
    const float* __restrict__ A, const float* __restrict__ Bm,
    const float* __restrict__ bias, float* __restrict__ C,
    int M, int N, int K, int relu)
{
    __shared__ __align__(16) __bf16 As[BM][LDK];
    __shared__ __align__(16) __bf16 BsT[BN][LDK];   // column-major B tile

    const int tid  = threadIdx.x;
    const int lane = tid & 31;
    const int wave = tid >> 5;       // 0..7
    const int wr   = wave & 3;       // row group (32 rows)
    const int wc   = wave >> 2;      // col group (64 cols)
    const int half = lane >> 4;
    const int l16  = lane & 15;

    const int rowBase = blockIdx.y * BM;
    const int colBase = blockIdx.x * BN;

    v8f acc[2][4] = {};

    const int kTiles = K / BK;
    for (int kt = 0; kt < kTiles; ++kt) {
        const int k0 = kt * BK;
        // ---- stage A tile 128x32: 16 contiguous floats/thread -> 2x b128 store
        {
            int r  = tid >> 1;
            int c0 = (tid & 1) * 16;
            const float* src = A + (size_t)(rowBase + r) * K + k0 + c0;
            *(uint4*)&As[r][c0]     = pk8(src);
            *(uint4*)&As[r][c0 + 8] = pk8(src + 8);
        }
        // ---- stage B tile 32x128 transposed: 2 rows x 8 cols per thread,
        //      packed (k,k+1) pairs -> 8 scattered ds_store_b32
        {
            int kk = (tid >> 4) * 2;        // even k
            int n0 = (tid & 15) * 8;
            const float* r0 = Bm + (size_t)(k0 + kk) * N + colBase + n0;
            const float* r1 = r0 + N;
            #pragma unroll
            for (int i = 0; i < 8; ++i)
                *(unsigned*)&BsT[n0 + i][kk] = pk2(r0[i], r1[i]);
        }
        if (kt + 1 < kTiles) {  // global_prefetch_b8
            __builtin_prefetch(A + (size_t)(rowBase + (tid >> 1)) * K + (kt + 1) * BK, 0, 1);
            __builtin_prefetch(Bm + (size_t)((kt + 1) * BK + (tid >> 4) * 2) * N + colBase + (tid & 15) * 8, 0, 1);
        }
        __syncthreads();

        const int kb = half * 8;
        v16bf a0 = ld_frag(&As[wr * 32 + l16][kb],      &As[wr * 32 + l16][16 + kb]);
        v16bf a1 = ld_frag(&As[wr * 32 + 16 + l16][kb], &As[wr * 32 + 16 + l16][16 + kb]);
        #pragma unroll
        for (int c = 0; c < 4; ++c) {
            const __bf16* bp = &BsT[wc * 64 + c * 16 + l16][half * 8];
            v16bf b = ld_frag(bp, bp + 16);
            acc[0][c] = __builtin_amdgcn_wmma_f32_16x16x32_bf16(
                false, a0, false, b, (short)0, acc[0][c], false, false);
            acc[1][c] = __builtin_amdgcn_wmma_f32_16x16x32_bf16(
                false, a1, false, b, (short)0, acc[1][c], false, false);
        }
        __syncthreads();
    }

    // epilogue
    #pragma unroll
    for (int c = 0; c < 4; ++c) {
        int   n  = colBase + wc * 64 + c * 16 + l16;
        float bv = bias ? bias[n] : 0.0f;
        #pragma unroll
        for (int sub = 0; sub < 2; ++sub) {
            #pragma unroll
            for (int r = 0; r < 8; ++r) {
                int   m = rowBase + wr * 32 + sub * 16 + r + half * 8;
                float v = acc[sub][c][r] + bv;
                if (relu) v = fmaxf(v, 0.0f);
                C[(size_t)m * N + n] = v;
            }
        }
    }
}

// =====================================================================
// Fused flash-attention: per (b,h), 4 waves x 16 q-rows per block.
// LDS layouts fragment-major: Ks[key][dh] (k-dim=dh), Vt[dh][key] (k-dim=key).
// =====================================================================
__global__ __launch_bounds__(128) void attn_kernel(
    const float* __restrict__ qkv,            // [B, L, 3D]
    const unsigned char* __restrict__ mask,   // [B, L] bool
    const float* __restrict__ rel_bias,       // [2*MAXREL+1, H]
    float* __restrict__ out)                  // [B, L, D]
{
    __shared__ __align__(16) __bf16 Ks[32][LDK];        // [key][dh]
    __shared__ __align__(16) __bf16 Vt[DH_][LDK];       // [dh][key]
    __shared__ __align__(16) __bf16 Pw[4][16][LDK];     // per-wave P scratch
    __shared__ float  maskf[32];
    __shared__ float  biasrow[2 * MAXREL_ + 1];

    const int tid  = threadIdx.x;
    const int lane = tid & 31;
    const int wave = tid >> 5;                // 0..3
    const int half = lane >> 4;
    const int l16  = lane & 15;

    const int bh = blockIdx.y;
    const int b  = bh / H_;
    const int h  = bh % H_;
    const int qbase = blockIdx.x * 64 + wave * 16;

    const float scale = 0.17677669529663687f;  // DH^-0.5

    if (tid < 2 * MAXREL_ + 1) biasrow[tid] = rel_bias[tid * H_ + h];

    // Q fragment (scale folded into bf16 conversion), A layout 16x32
    v16bf qa;
    {
        int m = qbase + l16;
        const float* qp = qkv + (size_t)(b * L_ + m) * (3 * D_) + h * DH_;
        int kb = half * 8;
        #pragma unroll
        for (int j = 0; j < 8; ++j) {
            int kk = ((j >= 4) ? 16 : 0) + kb + (j & 3) * 2;
            qa[2 * j]     = f2bf(qp[kk] * scale);
            qa[2 * j + 1] = f2bf(qp[kk + 1] * scale);
        }
    }

    v8f o0 = {}, o1 = {};
    float rmax[8], rsum[8];
    #pragma unroll
    for (int r = 0; r < 8; ++r) { rmax[r] = -1e30f; rsum[r] = 0.0f; }

    for (int kb0 = 0; kb0 < L_; kb0 += 32) {
        __syncthreads();  // previous-iter readers done before restage
        // ---- stage K natural: 8 contiguous dh per thread -> 1x b128 store
        {
            int kk = tid >> 2;
            int d0 = (tid & 3) * 8;
            const float* kp = qkv + (size_t)(b * L_ + kb0 + kk) * (3 * D_) + D_ + h * DH_ + d0;
            *(uint4*)&Ks[kk][d0] = pk8(kp);
        }
        // ---- stage V transposed: 2 keys x 4 dh per thread, packed pairs
        {
            int kk = (tid >> 3) * 2;        // even key
            int d0 = (tid & 7) * 4;
            const float* v0 = qkv + (size_t)(b * L_ + kb0 + kk) * (3 * D_) + 2 * D_ + h * DH_ + d0;
            const float* v1 = v0 + 3 * D_;
            #pragma unroll
            for (int i = 0; i < 4; ++i)
                *(unsigned*)&Vt[d0 + i][kk] = pk2(v0[i], v1[i]);
        }
        if (tid < 32) maskf[tid] = mask[(size_t)b * L_ + kb0 + tid] ? -1e9f : 0.0f;
        __syncthreads();

        // S = Q @ K^T : B-frag lane n=key, table-order k=dh
        const __bf16* kp0 = &Ks[l16][half * 8];
        const __bf16* kp1 = &Ks[16 + l16][half * 8];
        v16bf bk0 = ld_frag(kp0, kp0 + 16);
        v16bf bk1 = ld_frag(kp1, kp1 + 16);
        v8f s0 = {}, s1 = {};
        s0 = __builtin_amdgcn_wmma_f32_16x16x32_bf16(false, qa, false, bk0, (short)0, s0, false, false);
        s1 = __builtin_amdgcn_wmma_f32_16x16x32_bf16(false, qa, false, bk1, (short)0, s1, false, false);

        // bias + mask + online softmax
        const float mk0 = maskf[l16], mk1 = maskf[16 + l16];
        const int   j0 = kb0 + l16,  j1 = kb0 + 16 + l16;
        #pragma unroll
        for (int r = 0; r < 8; ++r) {
            int   qi  = qbase + r + half * 8;
            int   d0i = min(max(j0 - qi, -MAXREL_), MAXREL_) + MAXREL_;
            int   d1i = min(max(j1 - qi, -MAXREL_), MAXREL_) + MAXREL_;
            float v0  = s0[r] + biasrow[d0i] + mk0;
            float v1  = s1[r] + biasrow[d1i] + mk1;
            float t = fmaxf(v0, v1);
            t = fmaxf(t, __shfl_xor(t, 1, 32));
            t = fmaxf(t, __shfl_xor(t, 2, 32));
            t = fmaxf(t, __shfl_xor(t, 4, 32));
            t = fmaxf(t, __shfl_xor(t, 8, 32));
            float nm    = fmaxf(rmax[r], t);
            float alpha = __expf(rmax[r] - nm);
            rmax[r] = nm;
            float p0 = __expf(v0 - nm);
            float p1 = __expf(v1 - nm);
            float ps = p0 + p1;
            ps += __shfl_xor(ps, 1, 32);
            ps += __shfl_xor(ps, 2, 32);
            ps += __shfl_xor(ps, 4, 32);
            ps += __shfl_xor(ps, 8, 32);
            rsum[r] = rsum[r] * alpha + ps;
            o0[r] *= alpha;
            o1[r] *= alpha;
            int m = r + half * 8;
            Pw[wave][m][l16]      = f2bf(p0);
            Pw[wave][m][16 + l16] = f2bf(p1);
        }

        // repack P (C layout -> A layout) via per-wave LDS, then O += P @ V
        const int kb = half * 8;
        v16bf pa = ld_frag(&Pw[wave][l16][kb], &Pw[wave][l16][16 + kb]);
        const __bf16* vp0 = &Vt[l16][half * 8];
        const __bf16* vp1 = &Vt[16 + l16][half * 8];
        v16bf bv0 = ld_frag(vp0, vp0 + 16);
        v16bf bv1 = ld_frag(vp1, vp1 + 16);
        o0 = __builtin_amdgcn_wmma_f32_16x16x32_bf16(false, pa, false, bv0, (short)0, o0, false, false);
        o1 = __builtin_amdgcn_wmma_f32_16x16x32_bf16(false, pa, false, bv1, (short)0, o1, false, false);
    }

    // normalize and write out in [B, L, H*DH] layout
    #pragma unroll
    for (int r = 0; r < 8; ++r) {
        int    m   = qbase + r + half * 8;
        float  inv = 1.0f / rsum[r];
        float* op  = out + (size_t)(b * L_ + m) * D_ + h * DH_;
        op[l16]      = o0[r] * inv;
        op[16 + l16] = o1[r] * inv;
    }
}

// =====================================================================
// Fused residual-add + LayerNorm over D=256. 8 waves/block, 1 row/wave.
// =====================================================================
__global__ __launch_bounds__(256) void add_ln_kernel(
    const float* __restrict__ a, const float* __restrict__ bvec,
    const float* __restrict__ g, const float* __restrict__ beta,
    float* __restrict__ outp, int rows)
{
    int wave = threadIdx.x >> 5;
    int lane = threadIdx.x & 31;
    int row  = blockIdx.x * 8 + wave;
    if (row >= rows) return;
    const float* pa = a + (size_t)row * D_;
    const float* pb = bvec + (size_t)row * D_;
    float v[8];
    float s = 0.0f;
    #pragma unroll
    for (int i = 0; i < 8; ++i) {
        v[i] = pa[lane + 32 * i] + pb[lane + 32 * i];
        s += v[i];
    }
    #pragma unroll
    for (int off = 1; off < 32; off <<= 1) s += __shfl_xor(s, off, 32);
    float mu = s * (1.0f / D_);
    float q = 0.0f;
    #pragma unroll
    for (int i = 0; i < 8; ++i) { float d = v[i] - mu; q += d * d; }
    #pragma unroll
    for (int off = 1; off < 32; off <<= 1) q += __shfl_xor(q, off, 32);
    float rstd = rsqrtf(q * (1.0f / D_) + 1e-5f);
    float* po = outp + (size_t)row * D_;
    #pragma unroll
    for (int i = 0; i < 8; ++i) {
        int c = lane + 32 * i;
        po[c] = (v[i] - mu) * rstd * g[c] + beta[c];
    }
}

// =====================================================================
extern "C" void kernel_launch(void* const* d_in, const int* in_sizes, int n_in,
                              void* d_out, int out_size, void* d_ws, size_t ws_size,
                              hipStream_t stream) {
    (void)in_sizes; (void)n_in; (void)out_size; (void)ws_size;
    const float* x        = (const float*)d_in[0];
    const unsigned char* kpm = (const unsigned char*)d_in[1];
    const float* Wqkv     = (const float*)d_in[2];
    const float* bqkv     = (const float*)d_in[3];
    const float* Wo       = (const float*)d_in[4];
    const float* bo       = (const float*)d_in[5];
    const float* ln1_g    = (const float*)d_in[6];
    const float* ln1_b    = (const float*)d_in[7];
    const float* W1       = (const float*)d_in[8];
    const float* b1       = (const float*)d_in[9];
    const float* W2       = (const float*)d_in[10];
    const float* b2       = (const float*)d_in[11];
    const float* ln2_g    = (const float*)d_in[12];
    const float* ln2_b    = (const float*)d_in[13];
    const float* rel_bias = (const float*)d_in[14];
    float* y = (float*)d_out;

    // workspace carve-up
    float* qkv  = (float*)d_ws;                  // [8192, 768]
    float* attn = qkv  + (size_t)ROWS_ * 3 * D_; // [8192, 256]
    float* oprj = attn + (size_t)ROWS_ * D_;     // [8192, 256]
    float* x1   = oprj + (size_t)ROWS_ * D_;     // [8192, 256]
    float* m1   = x1   + (size_t)ROWS_ * D_;     // [8192, 1024]
    float* m2   = m1   + (size_t)ROWS_ * MLP_;   // [8192, 256]

    dim3 blk(256);
    // 1. QKV projection: [8192,256] @ [256,768]
    gemm_bf16<<<dim3(3 * D_ / BN, ROWS_ / BM), blk, 0, stream>>>(
        x, Wqkv, bqkv, qkv, ROWS_, 3 * D_, D_, 0);
    // 2. fused flash attention
    attn_kernel<<<dim3(L_ / 64, B_ * H_), dim3(128), 0, stream>>>(
        qkv, kpm, rel_bias, attn);
    // 3. output projection
    gemm_bf16<<<dim3(D_ / BN, ROWS_ / BM), blk, 0, stream>>>(
        attn, Wo, bo, oprj, ROWS_, D_, D_, 0);
    // 4. residual + LN1
    add_ln_kernel<<<dim3(ROWS_ / 8), blk, 0, stream>>>(
        x, oprj, ln1_g, ln1_b, x1, ROWS_);
    // 5. FC1 + ReLU: [8192,256] @ [256,1024]
    gemm_bf16<<<dim3(MLP_ / BN, ROWS_ / BM), blk, 0, stream>>>(
        x1, W1, b1, m1, ROWS_, MLP_, D_, 1);
    // 6. FC2: [8192,1024] @ [1024,256]
    gemm_bf16<<<dim3(D_ / BN, ROWS_ / BM), blk, 0, stream>>>(
        m1, W2, b2, m2, ROWS_, D_, MLP_, 0);
    // 7. residual + LN2 -> output
    add_ln_kernel<<<dim3(ROWS_ / 8), blk, 0, stream>>>(
        x1, m2, ln2_g, ln2_b, y, ROWS_);
}
